// Denoizer_25340307046554
// MI455X (gfx1250) — compile-verified, matrix operand order checked by reading, not run on hardware
//
#include <hip/hip_runtime.h>

// ---------------------------------------------------------------------------
// EGNN denoiser, fused for CDNA5 (gfx1250, wave32, WMMA 16x16x32 f16)
// ---------------------------------------------------------------------------

typedef _Float16 half_t;
typedef __attribute__((ext_vector_type(16))) _Float16 v16h;
typedef __attribute__((ext_vector_type(8)))  _Float16 v8h;
typedef __attribute__((ext_vector_type(8)))  float    v8f;

#define Bb 4
#define Nn 384
#define Ff 32
#define Hh 128
#define Ee 32

#define JSPLIT 8
#define JT_PER ((Nn / 16) / JSPLIT)   // 3 j-tiles per workgroup

// workspace layout (element offsets)
#define OFF_H    ((size_t)0)
#define OFF_P    (OFF_H   + (size_t)Bb * Nn * Hh)
#define OFF_Q    (OFF_P   + (size_t)Bb * Nn * Hh)
#define OFF_NA   (OFF_Q   + (size_t)Bb * Nn * Hh)
#define OFF_AGG  (OFF_NA  + (size_t)Bb * Nn * Hh)
#define OFF_ACCX (OFF_AGG + (size_t)Bb * Nn * Hh)
#define N_F32    (OFF_ACCX + (size_t)Bb * Nn * 3)
// half region (offsets in halves), 32B aligned (N_F32*4 % 32 == 0)
#define OFFH_WE1E ((size_t)0)          // 8 frags  * 512
#define OFFH_WE2  (OFFH_WE1E + 8  * 512)
#define OFFH_WC1  (OFFH_WE2  + 32 * 512)

#define NEG_LOG_1E4 (-9.210340371976184f)

// fast silu: x * v_rcp_f32(1 + exp(-x)) -- avoids the IEEE divide expansion
__device__ __forceinline__ float silu_f(float x) {
    return x * __builtin_amdgcn_rcpf(1.0f + __expf(-x));
}

// ---------------------------------------------------------------------------
// K0: pre-swizzle f16 weights into WMMA B-fragment order.
// Fragment (nt,ks): lane l, elem e holds W[k][n] with
//   k = ks*32 + (l>>4)*8 + (e&7) + ((e>>3)<<4),  n = nt*16 + (l&15)
// ---------------------------------------------------------------------------
__global__ __launch_bounds__(256) void prep_weights_kernel(
    const float* __restrict__ We1, const float* __restrict__ We2,
    const float* __restrict__ Wc1,
    half_t* __restrict__ fWe1e, half_t* __restrict__ fWe2,
    half_t* __restrict__ fWc1)
{
    int t = blockIdx.x * blockDim.x + threadIdx.x;
    if (t < 8 * 512) {  // We1 edge-attr rows (rows 2H+1 .. 2H+E), single k-step
        int frag = t >> 9, rem = t & 511, lane = rem >> 4, e = rem & 15;
        int k = ((lane >> 4) << 3) + (e & 7) + ((e >> 3) << 4);     // 0..31
        int n = frag * 16 + (lane & 15);
        fWe1e[t] = (half_t)We1[(size_t)(2 * Hh + 1 + k) * Hh + n];
    }
    if (t < 32 * 512) { // We2, Wc1 : 8 n-tiles x 4 k-steps
        int frag = t >> 9, rem = t & 511, lane = rem >> 4, e = rem & 15;
        int nt = frag >> 2, ks = frag & 3;
        int k = ks * 32 + ((lane >> 4) << 3) + (e & 7) + ((e >> 3) << 4);
        int n = nt * 16 + (lane & 15);
        fWe2[t] = (half_t)We2[(size_t)k * Hh + n];
        fWc1[t] = (half_t)Wc1[(size_t)k * Hh + n];
    }
}

// ---------------------------------------------------------------------------
// K1: per-node: h = feats@W_in + b_in ; node_attr ; P = h@We1a + be1 ; Q = h@We1b
// ---------------------------------------------------------------------------
__global__ __launch_bounds__(128) void node_embed_kernel(
    const float* __restrict__ feats, const int* __restrict__ idx,
    const float* __restrict__ W_in, const float* __restrict__ b_in,
    const float* __restrict__ We1, const float* __restrict__ be1,
    float* __restrict__ h, float* __restrict__ P, float* __restrict__ Q,
    float* __restrict__ nattr)
{
    int bn = blockIdx.x;
    int b = bn / Nn, n = bn % Nn;
    int c = threadIdx.x;

    __shared__ float s_f[Ff];
    __shared__ float s_h[Hh];
    if (c < Ff) s_f[c] = feats[(size_t)bn * Ff + c];
    __syncthreads();

    float acc = b_in[c];
#pragma unroll 4
    for (int f = 0; f < Ff; ++f) acc += s_f[f] * W_in[(size_t)f * Hh + c];
    s_h[c] = acc;
    h[(size_t)bn * Hh + c] = acc;

    // node_attr = sinusoid(n, H) + sinusoid(idx[b], H)
    float fq = __expf(NEG_LOG_1E4 * (float)(c & 63) * (1.0f / 64.0f));
    float aN = (float)n * fq;
    float aT = (float)idx[b] * fq;
    float na = (c < 64) ? (__sinf(aN) + __sinf(aT)) : (__cosf(aN) + __cosf(aT));
    nattr[(size_t)bn * Hh + c] = na;
    __syncthreads();

    float p = be1[c], q = 0.0f;
#pragma unroll 4
    for (int k = 0; k < Hh; ++k) {
        float hv = s_h[k];
        p += hv * We1[(size_t)k * Hh + c];
        q += hv * We1[(size_t)(Hh + k) * Hh + c];
    }
    P[(size_t)bn * Hh + c] = p;
    Q[(size_t)bn * Hh + c] = q;
}

// ---------------------------------------------------------------------------
// K2: fused all-pairs edge MLP + coordinate head, WMMA f16.
// Grid: Bb * (Nn/16) * JSPLIT blocks of 256 threads (8 wave32).
// Wave w handles M-tiles (i-rows) 2w, 2w+1; edge tile rows = j_local.
// ---------------------------------------------------------------------------
__global__ __launch_bounds__(256) void egnn_edge_kernel(
    const float* __restrict__ coords, const float* __restrict__ P,
    const float* __restrict__ Q, const float* __restrict__ We1,
    const float* __restrict__ Wc2, const float* __restrict__ be2,
    const float* __restrict__ bc1,
    const half_t* __restrict__ fWe1e, const half_t* __restrict__ fWe2,
    const half_t* __restrict__ fWc1,
    float* __restrict__ agg, float* __restrict__ accx)
{
    int wgid = blockIdx.x;
    int split = wgid % JSPLIT;
    int itile = (wgid / JSPLIT) % (Nn / 16);
    int b = wgid / (JSPLIT * (Nn / 16));
    int tid = threadIdx.x;
    int w = tid >> 5;
    int lane = tid & 31;
    int jl = lane & 15;
    int hiHalf = lane >> 4;

    __shared__ float s_Pi[16][Hh];
    __shared__ float s_Qj[16][Hh];
    __shared__ float s_xi[16][3];
    __shared__ float s_xj[16][3];
    __shared__ float s_rad[8][16];
    __shared__ float s_wr[Hh];
    __shared__ float s_wc2[Hh];
    __shared__ float s_be2[Hh];
    __shared__ float s_bc1[Hh];
    __shared__ __align__(16) half_t s_a[8][16][Hh];   // a1 / a3 ping
    __shared__ __align__(16) half_t s_m[8][16][Hh];   // m (f16) pong

    // i-tile data + small weights (once)
    for (int t = tid; t < 16 * Hh; t += 256) {
        int r = t >> 7, c = t & 127;
        s_Pi[r][c] = P[((size_t)(b * Nn + itile * 16 + r)) * Hh + c];
    }
    for (int t = tid; t < 48; t += 256)
        s_xi[t / 3][t % 3] = coords[((size_t)(b * Nn + itile * 16 + t / 3)) * 3 + (t % 3)];
    for (int t = tid; t < Hh; t += 256) {
        s_wr[t]  = We1[(size_t)(2 * Hh) * Hh + t];
        s_wc2[t] = Wc2[t];
        s_be2[t] = be2[t];
        s_bc1[t] = bc1[t];
    }
    __syncthreads();

    // per-lane sinusoid frequencies for edge embedding (E=32, half=16)
    float fr[8];
    int kbase = hiHalf * 8;
#pragma unroll
    for (int t = 0; t < 8; ++t)
        fr[t] = __expf(NEG_LOG_1E4 * (float)(kbase + t) * (1.0f / 16.0f));

    const v16h* bWe1e = (const v16h*)fWe1e;
    const v16h* bWe2  = (const v16h*)fWe2;
    const v16h* bWc1  = (const v16h*)fWc1;

    float aggp[2][8];
    float axp[2][3];
#pragma unroll
    for (int m2 = 0; m2 < 2; ++m2) {
#pragma unroll
        for (int n = 0; n < 8; ++n) aggp[m2][n] = 0.0f;
#pragma unroll
        for (int c = 0; c < 3; ++c) axp[m2][c] = 0.0f;
    }

    for (int jt = split * JT_PER; jt < split * JT_PER + JT_PER; ++jt) {
        // stage j-tile
        for (int t = tid; t < 16 * Hh; t += 256) {
            int r = t >> 7, c = t & 127;
            s_Qj[r][c] = Q[((size_t)(b * Nn + jt * 16 + r)) * Hh + c];
        }
        for (int t = tid; t < 48; t += 256)
            s_xj[t / 3][t % 3] = coords[((size_t)(b * Nn + jt * 16 + t / 3)) * 3 + (t % 3)];
        __syncthreads();

        for (int m2 = 0; m2 < 2; ++m2) {
            int iL = 2 * w + m2;
            int ig = itile * 16 + iL;      // global i
            int jdiag = ig - jt * 16;      // local j of the self-edge (may be OOR)

            // ---- geometry for edge (iL, jl) ----
            float dx = s_xi[iL][0] - s_xj[jl][0];
            float dy = s_xi[iL][1] - s_xj[jl][1];
            float dz = s_xi[iL][2] - s_xj[jl][2];
            float radial = dx * dx + dy * dy + dz * dz;
            float nrm = __builtin_sqrtf(radial + 1e-8f);
            float inv = __builtin_amdgcn_rcpf(nrm + 1.0f);
            float cdx = dx * inv, cdy = dy * inv, cdz = dz * inv;
            if (lane < 16) s_rad[w][lane] = radial;
            __builtin_amdgcn_wave_barrier();

            // ---- edge_attr A-fragment (16 edges x K=32, f16) ----
            v16h afrag;
#pragma unroll
            for (int e = 0; e < 8; ++e) {
                float ang = nrm * fr[e];
                afrag[e]     = (half_t)__sinf(ang);
                afrag[e + 8] = (half_t)__cosf(ang);
            }

            // ---- GEMM1: a1 = silu(P_i + Q_j + radial*wr + edge_attr@We1e) ----
#pragma unroll 1
            for (int nt = 0; nt < 8; ++nt) {
                int col = nt * 16 + jl;
                float pI = s_Pi[iL][col];
                float wrv = s_wr[col];
                v8f acc;
#pragma unroll
                for (int r = 0; r < 8; ++r) {
                    int jr = r + 8 * hiHalf;
                    acc[r] = pI + s_Qj[jr][col] + s_rad[w][jr] * wrv;
                }
                v16h bf = bWe1e[nt * 32 + lane];
                acc = __builtin_amdgcn_wmma_f32_16x16x32_f16(
                    false, afrag, false, bf, (short)0, acc, false, false);
#pragma unroll
                for (int r = 0; r < 8; ++r)
                    s_a[w][r + 8 * hiHalf][col] = (half_t)silu_f(acc[r]);
            }
            __builtin_amdgcn_wave_barrier();

            // ---- GEMM2: m = silu(a1 @ We2 + be2); accumulate agg partials ----
#pragma unroll 1
            for (int nt = 0; nt < 8; ++nt) {
                int col = nt * 16 + jl;
                float bias = s_be2[col];
                v8f acc;
#pragma unroll
                for (int r = 0; r < 8; ++r) acc[r] = bias;
#pragma unroll
                for (int ks = 0; ks < 4; ++ks) {
                    const v8h* ap = (const v8h*)&s_a[w][jl][ks * 32 + 8 * hiHalf];
                    v8h lo = ap[0];
                    v8h hi2 = ap[2];   // +16 halves
                    v16h af = __builtin_shufflevector(lo, hi2,
                        0, 1, 2, 3, 4, 5, 6, 7, 8, 9, 10, 11, 12, 13, 14, 15);
                    v16h bf = bWe2[(nt * 4 + ks) * 32 + lane];
                    acc = __builtin_amdgcn_wmma_f32_16x16x32_f16(
                        false, af, false, bf, (short)0, acc, false, false);
                }
                float s = 0.0f;
#pragma unroll
                for (int r = 0; r < 8; ++r) {
                    float v = silu_f(acc[r]);
                    int jr = r + 8 * hiHalf;
                    if (jr != jdiag) s += v;            // mask self-edge
                    s_m[w][jr][col] = (half_t)v;
                }
                aggp[m2][nt] += s;
            }
            __builtin_amdgcn_wave_barrier();

            // ---- GEMM3: a3 = silu(m @ Wc1 + bc1) -> s_a (reuse) ----
#pragma unroll 1
            for (int nt = 0; nt < 8; ++nt) {
                int col = nt * 16 + jl;
                float bias = s_bc1[col];
                v8f acc;
#pragma unroll
                for (int r = 0; r < 8; ++r) acc[r] = bias;
#pragma unroll
                for (int ks = 0; ks < 4; ++ks) {
                    const v8h* ap = (const v8h*)&s_m[w][jl][ks * 32 + 8 * hiHalf];
                    v8h lo = ap[0];
                    v8h hi2 = ap[2];
                    v16h af = __builtin_shufflevector(lo, hi2,
                        0, 1, 2, 3, 4, 5, 6, 7, 8, 9, 10, 11, 12, 13, 14, 15);
                    v16h bf = bWc1[(nt * 4 + ks) * 32 + lane];
                    acc = __builtin_amdgcn_wmma_f32_16x16x32_f16(
                        false, af, false, bf, (short)0, acc, false, false);
                }
#pragma unroll
                for (int r = 0; r < 8; ++r)
                    s_a[w][r + 8 * hiHalf][col] = (half_t)silu_f(acc[r]);
            }
            __builtin_amdgcn_wave_barrier();

            // ---- coordinate weight: cw = a3 . Wc2 (lane-pair split dot) ----
            float part = 0.0f;
            int k0 = 64 * hiHalf;
#pragma unroll 8
            for (int k = 0; k < 64; ++k)
                part += (float)s_a[w][jl][k0 + k] * s_wc2[k0 + k];
            float cw = part + __shfl_xor(part, 16);
            if (jt * 16 + jl == ig) cw = 0.0f;          // mask self-edge
            if (lane < 16) {
                axp[m2][0] += cdx * cw;
                axp[m2][1] += cdy * cw;
                axp[m2][2] += cdz * cw;
            }
        } // m2
        __syncthreads();   // protect s_Qj/s_xj before next j-tile
    } // jt

    // ---- epilogue: reduce + atomically accumulate agg / accx ----
#pragma unroll
    for (int m2 = 0; m2 < 2; ++m2) {
        int ig = itile * 16 + 2 * w + m2;
#pragma unroll
        for (int nt = 0; nt < 8; ++nt) {
            float v = aggp[m2][nt] + __shfl_xor(aggp[m2][nt], 16);
            if (lane < 16)
                atomicAdd(&agg[((size_t)(b * Nn + ig)) * Hh + nt * 16 + lane], v);
        }
        float ax = axp[m2][0], ay = axp[m2][1], az = axp[m2][2];
#pragma unroll
        for (int off = 8; off >= 1; off >>= 1) {
            ax += __shfl_xor(ax, off);
            ay += __shfl_xor(ay, off);
            az += __shfl_xor(az, off);
        }
        if (lane == 0) {
            atomicAdd(&accx[((size_t)(b * Nn + ig)) * 3 + 0], ax);
            atomicAdd(&accx[((size_t)(b * Nn + ig)) * 3 + 1], ay);
            atomicAdd(&accx[((size_t)(b * Nn + ig)) * 3 + 2], az);
        }
    }
}

// ---------------------------------------------------------------------------
// K3: node update + output heads
// ---------------------------------------------------------------------------
__global__ __launch_bounds__(128) void node_out_kernel(
    const float* __restrict__ feats, const float* __restrict__ h,
    const float* __restrict__ agg, const float* __restrict__ nattr,
    const float* __restrict__ Wn1, const float* __restrict__ bn1,
    const float* __restrict__ Wn2, const float* __restrict__ bn2,
    const float* __restrict__ W_out, const float* __restrict__ b_out,
    const float* __restrict__ accx, float* __restrict__ out)
{
    int bn = blockIdx.x;
    int c = threadIdx.x;
    __shared__ float s_cat[3 * Hh];
    __shared__ float s_u[Hh];
    __shared__ float s_h2[Hh];

    s_cat[c]          = h[(size_t)bn * Hh + c];
    s_cat[Hh + c]     = agg[(size_t)bn * Hh + c];
    s_cat[2 * Hh + c] = nattr[(size_t)bn * Hh + c];
    __syncthreads();

    float u = bn1[c];
#pragma unroll 4
    for (int k = 0; k < 3 * Hh; ++k) u += s_cat[k] * Wn1[(size_t)k * Hh + c];
    u = silu_f(u);
    s_u[c] = u;
    __syncthreads();

    float h2 = s_cat[c] + bn2[c];
#pragma unroll 4
    for (int k = 0; k < Hh; ++k) h2 += s_u[k] * Wn2[(size_t)k * Hh + c];
    s_h2[c] = h2;
    __syncthreads();

    if (c < Ff) {
        float o = b_out[c];
#pragma unroll 4
        for (int k = 0; k < Hh; ++k) o += s_h2[k] * W_out[(size_t)k * Ff + c];
        out[(size_t)(Bb * Nn * 3) + (size_t)bn * Ff + c] =
            o - feats[(size_t)bn * Ff + c];
    }
    if (c < 3)
        out[(size_t)bn * 3 + c] = accx[(size_t)bn * 3 + c];
}

// ---------------------------------------------------------------------------
extern "C" void kernel_launch(void* const* d_in, const int* in_sizes, int n_in,
                              void* d_out, int out_size, void* d_ws, size_t ws_size,
                              hipStream_t stream)
{
    (void)in_sizes; (void)n_in; (void)out_size; (void)ws_size;

    const float* coords = (const float*)d_in[0];
    const float* feats  = (const float*)d_in[1];
    const int*   idx    = (const int*)d_in[2];
    const float* W_in   = (const float*)d_in[3];
    const float* b_in   = (const float*)d_in[4];
    const float* We1    = (const float*)d_in[5];
    const float* be1    = (const float*)d_in[6];
    const float* We2    = (const float*)d_in[7];
    const float* be2    = (const float*)d_in[8];
    const float* Wc1    = (const float*)d_in[9];
    const float* bc1    = (const float*)d_in[10];
    const float* Wc2    = (const float*)d_in[11];
    const float* Wn1    = (const float*)d_in[12];
    const float* bn1    = (const float*)d_in[13];
    const float* Wn2    = (const float*)d_in[14];
    const float* bn2    = (const float*)d_in[15];
    const float* W_out  = (const float*)d_in[16];
    const float* b_out  = (const float*)d_in[17];

    float* out = (float*)d_out;
    float* wsF = (float*)d_ws;
    float* wh    = wsF + OFF_H;
    float* wP    = wsF + OFF_P;
    float* wQ    = wsF + OFF_Q;
    float* wNA   = wsF + OFF_NA;
    float* wAGG  = wsF + OFF_AGG;
    float* wACCX = wsF + OFF_ACCX;
    half_t* wsH  = (half_t*)((char*)d_ws + N_F32 * sizeof(float));
    half_t* fWe1e = wsH + OFFH_WE1E;
    half_t* fWe2  = wsH + OFFH_WE2;
    half_t* fWc1  = wsH + OFFH_WC1;

    // zero the accumulators (agg + accx are contiguous)
    hipMemsetAsync(wAGG, 0,
                   (size_t)(Bb * Nn * Hh + Bb * Nn * 3) * sizeof(float), stream);

    prep_weights_kernel<<<64, 256, 0, stream>>>(We1, We2, Wc1, fWe1e, fWe2, fWc1);

    node_embed_kernel<<<Bb * Nn, 128, 0, stream>>>(
        feats, idx, W_in, b_in, We1, be1, wh, wP, wQ, wNA);

    egnn_edge_kernel<<<Bb * (Nn / 16) * JSPLIT, 256, 0, stream>>>(
        coords, wP, wQ, We1, Wc2, be2, bc1, fWe1e, fWe2, fWc1, wAGG, wACCX);

    node_out_kernel<<<Bb * Nn, 128, 0, stream>>>(
        feats, wh, wAGG, wNA, Wn1, bn1, Wn2, bn2, W_out, b_out, wACCX, out);
}
